// Model_19104014532987
// MI455X (gfx1250) — matrix operand order repaired and checked by reading, not compiled, not run on hardware
//
#include <hip/hip_runtime.h>
#include <hip/hip_bf16.h>
#include <cstdint>
#include <cstddef>

// ---------------------------------------------------------------------------
// Model dims (fixed by the reference)
#define BB 8
#define TT 2048
#define EE 128          // EV == EKQ == 128
#define HH 4
#define DH 32           // head dim for both layers
#define MM (BB * TT)    // 16384 rows
#define FFD 512
#define OUTD 32

typedef _Float16 half16 __attribute__((ext_vector_type(16)));
typedef _Float16 half8  __attribute__((ext_vector_type(8)));
typedef float    float8 __attribute__((ext_vector_type(8)));

// ---------------------------------------------------------------------------
// WMMA fragment index helpers (16-bit operands, wave32; contiguous k order chosen by this code, not the ISA table order)
// A 16x32 (MxK): lane row M = lane%16; half h -> K = (lane>=16?16:0)+h
//                (same contiguous order as B).
// B 32x16 (KxN): lane col N = lane%16; half h -> K = (lane>=16?16:0)+h.
__device__ __forceinline__ int b_kidx(int lane, int h) {
  return ((lane & 16) ? 16 : 0) + h;
}
__device__ __forceinline__ half16 cat8(half8 lo, half8 hi) {
  return __builtin_shufflevector(lo, hi, 0, 1, 2, 3, 4, 5, 6, 7,
                                         8, 9, 10, 11, 12, 13, 14, 15);
}
// Load an A-fragment from a row-major row pointer (points at col kb of the row)
__device__ __forceinline__ half16 load_afrag(const _Float16* rowk, int o) {
  half8 lo = *(const half8*)(rowk + 2 * o);
  half8 hi = *(const half8*)(rowk + 2 * o + 8);
  return cat8(lo, hi);
}
// Load a B-fragment whose 16 halves are contiguous in memory at ptr
__device__ __forceinline__ half16 load_contig16(const _Float16* p) {
  half8 lo = *(const half8*)p;
  half8 hi = *(const half8*)(p + 8);
  return cat8(lo, hi);
}

// ---------------------------------------------------------------------------
// Elementwise converters
__global__ void k_f32_to_f16(const float* __restrict__ src,
                             _Float16* __restrict__ dst, int n) {
  int i = blockIdx.x * blockDim.x + threadIdx.x;
  if (i < n) dst[i] = (_Float16)src[i];
}

// X = concat(kq, v) along last dim (64+64 -> 128), converted to f16
__global__ void k_concat(const float* __restrict__ kq, const float* __restrict__ v,
                         _Float16* __restrict__ X) {
  int i = blockIdx.x * blockDim.x + threadIdx.x;
  if (i >= MM * 128) return;
  int r = i >> 7, c = i & 127;
  float s = (c < 64) ? kq[r * 64 + c] : v[r * 64 + (c - 64)];
  X[i] = (_Float16)s;
}

// Pre-swizzle a weight matrix W[KxN] (f32) into B-fragment lane-major order:
// Wp[((kc*tilesN + tn)*32 + lane)*16 + h] = W[(kc*32 + b_kidx(lane,h))*N + tn*16 + lane%16]
__global__ void k_pack_w(const float* __restrict__ W, _Float16* __restrict__ Wp,
                         int K, int N) {
  int i = blockIdx.x * blockDim.x + threadIdx.x;
  if (i >= K * N) return;
  const int h = i & 15;
  const int lane = (i >> 4) & 31;
  const int frag = i >> 9;            // kc*tilesN + tn
  const int tilesN = N >> 4;
  const int kc = frag / tilesN, tn = frag % tilesN;
  const int k = kc * 32 + b_kidx(lane, h);
  const int col = tn * 16 + (lane & 15);
  Wp[i] = (_Float16)W[(size_t)k * N + col];
}

// ---------------------------------------------------------------------------
// WMMA GEMM: C = act(A[MxK] * W[KxN] + bias), W pre-swizzled by k_pack_w.
// One wave computes a 32 x (16*NB) tile: 2 A-frags x NB B-frags = 2*NB WMMAs/K-step.
// act: 0 = none, 1 = exact GELU. C32/C16/C16t each optional (nullable).
// C16t (only valid for N==128, M==MM) writes the transposed-head layout
// Vt[(b*128 + col)*T + t] used by the attention kernel's V fragments.
template <int NB>
__global__ void k_gemm(const _Float16* __restrict__ A, const _Float16* __restrict__ Wp,
                       const float* __restrict__ bias,
                       float* __restrict__ C32, _Float16* __restrict__ C16,
                       _Float16* __restrict__ C16t,
                       int M, int N, int K, int act) {
  const int lane = threadIdx.x & 31;
  const int wave = threadIdx.x >> 5;
  const int tilesN = N >> 4;
  const int nbt = tilesN / NB;
  const int tilesTotal = (M >> 5) * nbt;
  const int tile = blockIdx.x * (blockDim.x >> 5) + wave;
  if (tile >= tilesTotal) return;                 // wave-uniform exit
  const int tm = tile / nbt, tn0 = (tile % nbt) * NB;
  const int r16 = lane & 15;
  const int hi  = lane >> 4;
  const int o   = hi ? 8 : 0;

  float8 acc[2][NB] = {};
  const _Float16* arow0 = A + (size_t)(tm * 32 + r16) * K;
  const _Float16* arow1 = arow0 + (size_t)16 * K;

  for (int kb = 0, kc = 0; kb < K; kb += 32, ++kc) {
    const half16 a0 = load_afrag(arow0 + kb, o);
    const half16 a1 = load_afrag(arow1 + kb, o);
#pragma unroll
    for (int f = 0; f < NB; ++f) {
      const _Float16* wp = Wp + (((size_t)kc * tilesN + tn0 + f) * 32 + lane) * 16;
      const half16 b = load_contig16(wp);
      acc[0][f] = __builtin_amdgcn_wmma_f32_16x16x32_f16(false, a0, false, b,
                                                         (short)0, acc[0][f], false, false);
      acc[1][f] = __builtin_amdgcn_wmma_f32_16x16x32_f16(false, a1, false, b,
                                                         (short)0, acc[1][f], false, false);
    }
  }

#pragma unroll
  for (int mi = 0; mi < 2; ++mi) {
#pragma unroll
    for (int f = 0; f < NB; ++f) {
      const int ocol = (tn0 + f) * 16 + r16;
      const float bv = bias ? bias[ocol] : 0.0f;
      const int row0 = tm * 32 + mi * 16 + hi * 8;
#pragma unroll
      for (int r = 0; r < 8; ++r) {
        float x = acc[mi][f][r] + bv;
        if (act == 1) x = 0.5f * x * (1.0f + erff(x * 0.70710678118654752f));
        const int row = row0 + r;
        const size_t off = (size_t)row * N + ocol;
        if (C32) C32[off] = x;
        if (C16) C16[off] = (_Float16)x;
        if (C16t) {
          const int b = row >> 11;          // row / T
          const int t = row & (TT - 1);     // row % T
          C16t[((size_t)(b * 128 + ocol)) * TT + t] = (_Float16)x;
        }
      }
    }
  }
}

// ---------------------------------------------------------------------------
// Flash attention with WMMA.
// Q,K: [B*T, 128] f16, heads in cols [h*32, h*32+32).
// Vt:  transposed f16, Vt[(b*128 + h*32 + d)*T + t].
// One wave per (b, head, 16-query tile); 32-key chunks, online softmax.
// mode 0: causal j<=i (layer A); mode 1: strict j<i plus (0,0) (layer B).
// No 1/sqrt(d) scaling (faithful to reference).
__global__ void k_attn(const _Float16* __restrict__ Q, const _Float16* __restrict__ Kx,
                       const _Float16* __restrict__ Vt,
                       float* __restrict__ O32, _Float16* __restrict__ O16, int mode) {
  const int lane = threadIdx.x & 31;
  const int wave = threadIdx.x >> 5;
  const int gw = blockIdx.x * 4 + wave;   // 0 .. B*H*(T/16)-1
  const int qt = gw & 127;
  const int hd = (gw >> 7) & 3;
  const int b  = gw >> 9;

  __shared__ _Float16 pbuf[4][16][32];    // per-wave P staging (D-layout -> A-layout)

  const int r16 = lane & 15;
  const int hi  = lane >> 4;
  const int o   = hi ? 8 : 0;
  const size_t base  = ((size_t)b * TT) * EE + (size_t)hd * DH;
  const size_t vbase = ((size_t)(b * 128 + hd * DH)) * TT;

  // Q fragment (constant over the key loop): 2 x b128
  const half16 qf = load_afrag(Q + base + (size_t)(qt * 16 + r16) * EE, o);

  float8 o1 = {}, o2 = {};
  float mrow[8], lrow[8];
#pragma unroll
  for (int r = 0; r < 8; ++r) { mrow[r] = -1e30f; lrow[r] = 0.0f; }

  const int jmax = (mode == 0) ? (qt * 16 + 15)
                               : ((qt == 0) ? 0 : (qt * 16 + 14));
  const int nch = jmax / 32 + 1;

  for (int c = 0; c < nch; ++c) {
    const int kb = c * 32;

    // ---- S = Q * K^T for 32 keys (two 16-key WMMAs); K rows contiguous in d ----
    const half16 bk1 = load_contig16(Kx + base + (size_t)(kb + r16) * EE + hi * 16);
    const half16 bk2 = load_contig16(Kx + base + (size_t)(kb + 16 + r16) * EE + hi * 16);
    if (c + 1 < nch) {   // prefetch hint for next K chunk
      __builtin_prefetch(&Kx[base + (size_t)(kb + 32 + r16) * EE], 0, 1);
    }
    const float8 z = {};
    float8 s1 = __builtin_amdgcn_wmma_f32_16x16x32_f16(false, qf, false, bk1,
                                                       (short)0, z, false, false);
    float8 s2 = __builtin_amdgcn_wmma_f32_16x16x32_f16(false, qf, false, bk2,
                                                       (short)0, z, false, false);

    // ---- mask + online softmax (row stats live in 16-lane halves) ----
#pragma unroll
    for (int r = 0; r < 8; ++r) {
      const int i  = qt * 16 + hi * 8 + r;
      const int j1 = kb + r16;
      const int j2 = j1 + 16;
      float v1 = s1[r], v2 = s2[r];
      bool ok1, ok2;
      if (mode == 0) { ok1 = (j1 <= i); ok2 = (j2 <= i); }
      else { ok1 = (j1 < i) || (i == 0 && j1 == 0); ok2 = (j2 < i); }
      if (!ok1) v1 = -1e30f;
      if (!ok2) v2 = -1e30f;

      float mc = fmaxf(v1, v2);
      mc = fmaxf(mc, __shfl_xor(mc, 1, 32));
      mc = fmaxf(mc, __shfl_xor(mc, 2, 32));
      mc = fmaxf(mc, __shfl_xor(mc, 4, 32));
      mc = fmaxf(mc, __shfl_xor(mc, 8, 32));
      const float mn = fmaxf(mrow[r], mc);
      const float scale = __expf(mrow[r] - mn);
      const float p1 = __expf(v1 - mn);
      const float p2 = __expf(v2 - mn);
      float ps = p1 + p2;
      ps += __shfl_xor(ps, 1, 32);
      ps += __shfl_xor(ps, 2, 32);
      ps += __shfl_xor(ps, 4, 32);
      ps += __shfl_xor(ps, 8, 32);
      lrow[r] = lrow[r] * scale + ps;
      mrow[r] = mn;
      o1[r] *= scale;
      o2[r] *= scale;

      pbuf[wave][hi * 8 + r][r16]      = (_Float16)p1;
      pbuf[wave][hi * 8 + r][r16 + 16] = (_Float16)p2;
    }
    asm volatile("s_wait_dscnt 0" ::: "memory");

    // ---- O += P * V : P via 2 x ds_load_b128, V via transposed contiguous rows ----
    const half16 pf = load_afrag(&pbuf[wave][r16][0], o);
    const half16 bv1 = load_contig16(Vt + vbase + (size_t)r16 * TT + kb + hi * 16);
    const half16 bv2 = load_contig16(Vt + vbase + (size_t)(16 + r16) * TT + kb + hi * 16);
    o1 = __builtin_amdgcn_wmma_f32_16x16x32_f16(false, pf, false, bv1,
                                                (short)0, o1, false, false);
    o2 = __builtin_amdgcn_wmma_f32_16x16x32_f16(false, pf, false, bv2,
                                                (short)0, o2, false, false);
  }

  // ---- finalize: divide by row sums, write out (heads re-interleaved) ----
#pragma unroll
  for (int r = 0; r < 8; ++r) {
    const float inv = 1.0f / lrow[r];
    const int t = qt * 16 + hi * 8 + r;
    const size_t off = base + (size_t)t * EE;
    const float x1 = o1[r] * inv;
    const float x2 = o2[r] * inv;
    if (O32) { O32[off + r16] = x1; O32[off + 16 + r16] = x2; }
    if (O16) { O16[off + r16] = (_Float16)x1; O16[off + 16 + r16] = (_Float16)x2; }
  }
}

// ---------------------------------------------------------------------------
// Fused residual + LayerNorm over 128 cols: out = LN(a + b) * g + be
__global__ void k_add_ln(const float* __restrict__ a, const float* __restrict__ bsrc,
                         const float* __restrict__ g, const float* __restrict__ be,
                         float* __restrict__ out32, _Float16* __restrict__ out16) {
  const int row = blockIdx.x;
  const int c = threadIdx.x;       // 128 threads
  __shared__ float red[128];
  const size_t off = (size_t)row * 128 + c;
  const float x = a[off] + bsrc[off];

  red[c] = x; __syncthreads();
  for (int s = 64; s > 0; s >>= 1) { if (c < s) red[c] += red[c + s]; __syncthreads(); }
  const float mean = red[0] * (1.0f / 128.0f);
  __syncthreads();

  const float d = x - mean;
  red[c] = d * d; __syncthreads();
  for (int s = 64; s > 0; s >>= 1) { if (c < s) red[c] += red[c + s]; __syncthreads(); }
  const float var = red[0] * (1.0f / 128.0f);

  const float y = d * rsqrtf(var + 1e-5f) * g[c] + be[c];
  if (out32) out32[off] = y;
  if (out16) out16[off] = (_Float16)y;
}

// ---------------------------------------------------------------------------
extern "C" void kernel_launch(void* const* d_in, const int* in_sizes, int n_in,
                              void* d_out, int out_size, void* d_ws, size_t ws_size,
                              hipStream_t stream) {
  (void)in_sizes; (void)n_in; (void)out_size; (void)ws_size;
  const float* kq    = (const float*)d_in[0];
  const float* v     = (const float*)d_in[1];
  const float* A_Wk  = (const float*)d_in[2];
  const float* A_bk  = (const float*)d_in[3];
  const float* A_Wq  = (const float*)d_in[4];
  const float* A_bq  = (const float*)d_in[5];
  const float* A_Wv  = (const float*)d_in[6];
  const float* A_bv  = (const float*)d_in[7];
  const float* A_f1W = (const float*)d_in[8];
  const float* A_f1b = (const float*)d_in[9];
  const float* A_f2W = (const float*)d_in[10];
  const float* A_f2b = (const float*)d_in[11];
  const float* A_n1g = (const float*)d_in[12];
  const float* A_n1b = (const float*)d_in[13];
  const float* A_n2g = (const float*)d_in[14];
  const float* A_n2b = (const float*)d_in[15];
  const float* B_Wk  = (const float*)d_in[16];
  const float* B_bk  = (const float*)d_in[17];
  const float* B_Wq  = (const float*)d_in[18];
  const float* B_bq  = (const float*)d_in[19];
  const float* B_Wv  = (const float*)d_in[20];
  const float* B_bv  = (const float*)d_in[21];
  const float* B_f1W = (const float*)d_in[22];
  const float* B_f1b = (const float*)d_in[23];
  const float* B_f2W = (const float*)d_in[24];
  const float* B_f2b = (const float*)d_in[25];
  const float* out_W = (const float*)d_in[26];
  const float* out_b = (const float*)d_in[27];
  float* out = (float*)d_out;

  char* ws = (char*)d_ws;
  size_t cur = 0;
  auto alloc = [&](size_t bytes) -> void* {
    void* p = ws + cur;
    cur = (cur + bytes + 255) & ~(size_t)255;
    return p;
  };

  // ---- workspace layout (aliased across phases) ----
  _Float16* Xh   = (_Float16*)alloc((size_t)MM * 128 * 2);  // later: AO2h
  _Float16* kqh  = (_Float16*)alloc((size_t)MM * 64 * 2);
  _Float16* Qh   = (_Float16*)alloc((size_t)MM * 128 * 2);  // QA then Q2
  _Float16* Kh   = (_Float16*)alloc((size_t)MM * 128 * 2);  // KA then K2
  _Float16* Vt   = (_Float16*)alloc((size_t)MM * 128 * 2);  // transposed V (A then B)
  float*    VA32 = (float*)   alloc((size_t)MM * 128 * 4);  // later: vAh (f16)
  float*    AO32 = (float*)   alloc((size_t)MM * 128 * 4);  // later: FF32
  float*    h132 = (float*)   alloc((size_t)MM * 128 * 4);
  _Float16* h1h  = (_Float16*)alloc((size_t)MM * 128 * 2);  // later: FF2h
  _Float16* G1h  = (_Float16*)alloc((size_t)MM * FFD * 2);  // later: G2h
  // pre-swizzled f16 weights
  _Float16* wAk  = (_Float16*)alloc(128 * 128 * 2);
  _Float16* wAq  = (_Float16*)alloc(128 * 128 * 2);
  _Float16* wAv  = (_Float16*)alloc(128 * 128 * 2);
  _Float16* wAf1 = (_Float16*)alloc(128 * FFD * 2);
  _Float16* wAf2 = (_Float16*)alloc(FFD * 128 * 2);
  _Float16* wBk  = (_Float16*)alloc(64 * 128 * 2);
  _Float16* wBq  = (_Float16*)alloc(64 * 128 * 2);
  _Float16* wBv  = (_Float16*)alloc(128 * 128 * 2);
  _Float16* wBf1 = (_Float16*)alloc(128 * FFD * 2);
  _Float16* wBf2 = (_Float16*)alloc(FFD * 128 * 2);
  _Float16* wOut = (_Float16*)alloc(128 * OUTD * 2);
  // aliases
  _Float16* AO2h = Xh;
  _Float16* vAh  = (_Float16*)VA32;
  float*    FF32 = AO32;
  _Float16* FF2h = h1h;
  _Float16* G2h  = G1h;

  auto pack = [&](const float* src, _Float16* dst, int K, int N) {
    k_pack_w<<<(K * N + 255) / 256, 256, 0, stream>>>(src, dst, K, N);
  };
  auto gemm = [&](const _Float16* A, const _Float16* Wp, const float* bias,
                  float* C32, _Float16* C16, _Float16* C16t,
                  int M, int N, int K, int act) {
    const int tilesN = N / 16;
    if (tilesN % 4 == 0) {
      int tiles = (M / 32) * (tilesN / 4);
      k_gemm<4><<<(tiles + 7) / 8, 256, 0, stream>>>(A, Wp, bias, C32, C16, C16t, M, N, K, act);
    } else {
      int tiles = (M / 32) * (tilesN / 2);
      k_gemm<2><<<(tiles + 7) / 8, 256, 0, stream>>>(A, Wp, bias, C32, C16, C16t, M, N, K, act);
    }
  };

  // ---- weight packing + input conversion ----
  pack(A_Wk, wAk, 128, 128);   pack(A_Wq, wAq, 128, 128);  pack(A_Wv, wAv, 128, 128);
  pack(A_f1W, wAf1, 128, FFD); pack(A_f2W, wAf2, FFD, 128);
  pack(B_Wk, wBk, 64, 128);    pack(B_Wq, wBq, 64, 128);   pack(B_Wv, wBv, 128, 128);
  pack(B_f1W, wBf1, 128, FFD); pack(B_f2W, wBf2, FFD, 128);
  pack(out_W, wOut, 128, OUTD);
  k_f32_to_f16<<<(MM * 64 + 255) / 256, 256, 0, stream>>>(kq, kqh, MM * 64);
  k_concat<<<(MM * 128 + 255) / 256, 256, 0, stream>>>(kq, v, Xh);

  // ---- Layer A ----
  gemm(Xh, wAk, A_bk, nullptr, Kh, nullptr, MM, 128, 128, 0);
  gemm(Xh, wAq, A_bq, nullptr, Qh, nullptr, MM, 128, 128, 0);
  gemm(Xh, wAv, A_bv, VA32, nullptr, Vt, MM, 128, 128, 0);   // V: f32 + transposed f16
  k_attn<<<(BB * HH * (TT / 16)) / 4, 128, 0, stream>>>(Qh, Kh, Vt, AO32, nullptr, 0);
  k_add_ln<<<MM, 128, 0, stream>>>(VA32, AO32, A_n1g, A_n1b, h132, h1h);
  gemm(h1h, wAf1, A_f1b, nullptr, G1h, nullptr, MM, FFD, 128, 1);   // GELU
  gemm(G1h, wAf2, A_f2b, FF32, nullptr, nullptr, MM, 128, FFD, 0);
  k_add_ln<<<MM, 128, 0, stream>>>(h132, FF32, A_n2g, A_n2b, nullptr, vAh);

  // ---- Layer B ----
  gemm(kqh, wBk, B_bk, nullptr, Kh, nullptr, MM, 128, 64, 0);
  gemm(kqh, wBq, B_bq, nullptr, Qh, nullptr, MM, 128, 64, 0);
  gemm(vAh, wBv, B_bv, nullptr, nullptr, Vt, MM, 128, 128, 0);  // V2 -> transposed only
  k_attn<<<(BB * HH * (TT / 16)) / 4, 128, 0, stream>>>(Qh, Kh, Vt, nullptr, AO2h, 1);
  gemm(AO2h, wBf1, B_f1b, nullptr, G2h, nullptr, MM, FFD, 128, 1);  // GELU
  gemm(G2h, wBf2, B_f2b, nullptr, FF2h, nullptr, MM, 128, FFD, 0);

  // ---- Output head ----
  gemm(FF2h, wOut, out_b, out, nullptr, nullptr, MM, OUTD, 128, 0);
}
